// LSHAttention_82867099009347
// MI455X (gfx1250) — matrix operand order repaired and checked by reading, not compiled, not run on hardware
//
#include <hip/hip_runtime.h>

typedef __attribute__((ext_vector_type(16))) _Float16 v16h;
typedef __attribute__((ext_vector_type(8)))  float    v8f;

// Problem constants (from the reference)
constexpr int kBatch = 2;
constexpr int kN     = 4096;
constexpr int kD     = 1024;
constexpr int kH     = 16;
constexpr int kR     = 4;
constexpr int kM     = 8;
constexpr int kBkt   = 64;           // bucket size
constexpr int kDh    = kD / kH;      // 64 head dim
constexpr int kC     = kN / kBkt;    // 64 buckets
constexpr int kPit   = kDh + 8;      // f16 LDS pitch (72 halves = 144B, conflict-skewed)

// ---------------------------------------------------------------------------
// scale_i = clamp(int(1 + vscale * mean(valence)), 1, M)   (trunc toward zero)
// ---------------------------------------------------------------------------
__device__ __forceinline__ int lsh_scale_i(const float* valence, const float* vscale) {
  float mean = 0.f;
#pragma unroll
  for (int i = 0; i < kBatch; ++i) mean += valence[i];
  mean *= (1.0f / kBatch);
  float s = 1.0f + vscale[0] * mean;
  int si = (int)s;
  if (si < 1) si = 1;
  if (si > kM) si = kM;
  return si;
}

// ---------------------------------------------------------------------------
// Kernel 0: zero the attn-mean accumulation region
// ---------------------------------------------------------------------------
__global__ void lsh_zero_kernel(float* __restrict__ p, int n) {
  int i = blockIdx.x * blockDim.x + threadIdx.x;
  if (i < n) p[i] = 0.f;
}

// ---------------------------------------------------------------------------
// Kernel 1: LSH codes.  One block per (b,r,h); 64x8 projection tile in LDS.
// codes[(b,r,h,n)] = argmax_m( Q[b,n,h,:] . proj[r,h,:,m] ) * scale_i
// ---------------------------------------------------------------------------
__global__ void __launch_bounds__(256)
lsh_codes_kernel(const float* __restrict__ q, const float* __restrict__ proj,
                 const float* __restrict__ valence, const float* __restrict__ vscale,
                 int* __restrict__ codes) {
  __shared__ float pj[kDh][kM];
  const int bid = blockIdx.x;
  const int h = bid % kH;
  const int r = (bid / kH) % kR;
  const int b = bid / (kH * kR);

  for (int e = threadIdx.x; e < kDh * kM; e += blockDim.x) {
    int k = e / kM, m = e % kM;
    pj[k][m] = proj[((size_t)(r * kH + h) * kDh + k) * kM + m];
  }
  __syncthreads();

  const int si = lsh_scale_i(valence, vscale);
  int* cb = codes + ((size_t)(b * kR + r) * kH + h) * kN;

  for (int n = threadIdx.x; n < kN; n += blockDim.x) {
    const float* qrow = q + (size_t)(b * kN + n) * kD + h * kDh;
    float acc[kM];
#pragma unroll
    for (int m = 0; m < kM; ++m) acc[m] = 0.f;
#pragma unroll 4
    for (int k4 = 0; k4 < kDh; k4 += 4) {
      const float4 qv = *(const float4*)(qrow + k4);   // global_load_b128
#pragma unroll
      for (int m = 0; m < kM; ++m) {
        acc[m] += qv.x * pj[k4 + 0][m];
        acc[m] += qv.y * pj[k4 + 1][m];
        acc[m] += qv.z * pj[k4 + 2][m];
        acc[m] += qv.w * pj[k4 + 3][m];
      }
    }
    // first-max argmax (matches jnp.argmax tie-break)
    float best = acc[0];
    int bi = 0;
#pragma unroll
    for (int m = 1; m < kM; ++m)
      if (acc[m] > best) { best = acc[m]; bi = m; }
    cb[n] = bi * si;
  }
}

// ---------------------------------------------------------------------------
// Kernel 2: stable counting sort (== stable jnp.argsort for 8-valued keys).
// One wave32 per (b,r,h).  Phase 1: LDS histogram.  Phase 2: ballot-ranked
// stable scatter: rank within chunk = popc(ballot(code==bin) & lanemask_lt).
// ---------------------------------------------------------------------------
__global__ void __launch_bounds__(32)
lsh_sort_kernel(const int* __restrict__ codes, int* __restrict__ idx,
                const float* __restrict__ valence, const float* __restrict__ vscale) {
  __shared__ int hist[kM];
  __shared__ int offs[kM];
  const int bid  = blockIdx.x;
  const int lane = threadIdx.x;           // wave32
  const int si   = lsh_scale_i(valence, vscale);
  const int* cb  = codes + (size_t)bid * kN;
  int* ob        = idx   + (size_t)bid * kN;

  if (lane < kM) hist[lane] = 0;
  __builtin_amdgcn_wave_barrier();
  for (int n = lane; n < kN; n += 32)
    atomicAdd(&hist[cb[n] / si], 1);      // code = bin * si, si >= 1
  __builtin_amdgcn_wave_barrier();
  if (lane == 0) {
    int run = 0;
#pragma unroll
    for (int m = 0; m < kM; ++m) { offs[m] = run; run += hist[m]; }
  }
  __builtin_amdgcn_wave_barrier();

  const unsigned lt = (1u << lane) - 1u;
  for (int base = 0; base < kN; base += 32) {
    const int bin = cb[base + lane] / si;
#pragma unroll
    for (int m = 0; m < kM; ++m) {
      unsigned mask = (unsigned)__ballot(bin == m);
      if (bin == m) {
        int pos = offs[m] + __popc(mask & lt);
        ob[pos] = base + lane;
      }
      __builtin_amdgcn_wave_barrier();
      if (lane == 0) offs[m] += __popc(mask);
      __builtin_amdgcn_wave_barrier();
    }
  }
}

// ---------------------------------------------------------------------------
// WMMA fragment loaders (contiguous k = 16*lhalf + s order, wave32).  All loaders read
// CONTIGUOUS f16 pairs so they lower to merged ds_load_b128.
// ---------------------------------------------------------------------------
// A (16x32 f16): M = row0 + lane%16; VGPR p -> K = 16*lhalf + 2p{,+1}
__device__ __forceinline__ v16h load_a_frag(const _Float16 (*S)[kPit],
                                            int row0, int k0, int lhalf, int ln) {
  v16h a;
  const int m = row0 + ln;
#pragma unroll
  for (int p = 0; p < 8; ++p) {
    const int kk = lhalf * 16 + 2 * p;
    a[2 * p]     = S[m][k0 + kk];
    a[2 * p + 1] = S[m][k0 + kk + 1];
  }
  return a;
}

// B (32x16 f16, KxN) where B[k][n] = S[col0+n][k].  Used both for Q.K^T
// (S = Ks, row-major keys) and for P.V (S = VsT, d-major transposed V).
// lane: n = lane%16; lanes 0-15 hold K=0..15, lanes 16-31 hold K=16..31, 2 K/VGPR
__device__ __forceinline__ v16h load_bT_frag(const _Float16 (*S)[kPit],
                                             int col0, int k0, int lhalf, int ln) {
  v16h bb;
  const int rrow = col0 + ln;
  const int kb0  = k0 + lhalf * 16;
#pragma unroll
  for (int p = 0; p < 8; ++p) {
    bb[2 * p]     = S[rrow][kb0 + 2 * p];
    bb[2 * p + 1] = S[rrow][kb0 + 2 * p + 1];
  }
  return bb;
}

// ---------------------------------------------------------------------------
// Kernel 3: bucketed attention.  One block per (b,h,c), 256 threads = 8 waves.
// Each wave owns tiles (tiq, tjq) and (tiq+2, tjq) of the 4x4 16x16 tile grid
// -- both share the SAME column group, so every GEMM's B-fragment is loaded
// once and fed to two WMMAs.  Q A-fragments are hoisted out of the round loop.
// PV accumulation over rounds rides in the WMMA C operand.
// ---------------------------------------------------------------------------
__global__ void __launch_bounds__(256)
lsh_attn_kernel(const float* __restrict__ q, const float* __restrict__ key,
                const float* __restrict__ val, const int* __restrict__ idx,
                float* __restrict__ out, float* __restrict__ attn_out) {
  __shared__ _Float16 Qs[kBkt][kPit];        // Q rows (s x d)
  __shared__ _Float16 Ks[kBkt][kPit];        // gathered K rows (e x d)
  __shared__ _Float16 VsT[kDh][kPit];        // gathered V TRANSPOSED (d x e)
  __shared__ _Float16 Ps[kBkt][kPit];        // probabilities (s x e)
  __shared__ float    Ss[kBkt][kBkt + 1];
  __shared__ float    Aacc[kBkt][kBkt + 1];
  __shared__ int      sidx[kBkt];            // staged bucket indices

  const int bid  = blockIdx.x;
  const int c    = bid % kC;
  const int h    = (bid / kC) % kH;
  const int b    = bid / (kC * kH);
  const int tid  = threadIdx.x;
  const int wave = tid >> 5;
  const int lane = tid & 31;
  const int lhalf = lane >> 4;     // 0: lanes 0-15, 1: lanes 16-31
  const int ln    = lane & 15;
  const int tiq   = wave >> 2;     // tile rows tiq and tiq+2
  const int tjq   = wave & 3;      // shared tile column

  // Q bucket in ORIGINAL order (faithful to reference), f32 -> f16, b128 loads
  for (int e = tid; e < kBkt * kDh / 4; e += 256) {
    const int row = e >> 4, c4 = (e & 15) * 4;
    const float4 qv = *(const float4*)(q + (size_t)(b * kN + c * kBkt + row) * kD
                                         + h * kDh + c4);
    Qs[row][c4 + 0] = (_Float16)qv.x;
    Qs[row][c4 + 1] = (_Float16)qv.y;
    Qs[row][c4 + 2] = (_Float16)qv.z;
    Qs[row][c4 + 3] = (_Float16)qv.w;
  }
  for (int e = tid; e < kBkt * kBkt; e += 256) Aacc[e >> 6][e & 63] = 0.f;
  __syncthreads();

  // Hoist Q A-fragments (constant across rounds): [tile-half][k-chunk]
  v16h qa[2][2];
  qa[0][0] = load_a_frag(Qs, tiq * 16,        0, lhalf, ln);
  qa[0][1] = load_a_frag(Qs, tiq * 16,       32, lhalf, ln);
  qa[1][0] = load_a_frag(Qs, (tiq + 2) * 16,  0, lhalf, ln);
  qa[1][1] = load_a_frag(Qs, (tiq + 2) * 16, 32, lhalf, ln);

  v8f oacc0 = {}, oacc1 = {};

  for (int r = 0; r < kR; ++r) {
    // stage bucket indices into LDS (kills per-row dependent global loads)
    const int* ib = idx + ((size_t)(b * kR + r) * kH + h) * kN + c * kBkt;
    if (tid < kBkt) sidx[tid] = ib[tid];
    // prefetch next round's bucket index list (global_prefetch_b8 path)
    if (r + 1 < kR && tid >= kBkt && tid < kBkt + 4) {
      const int* ibn = idx + ((size_t)(b * kR + r + 1) * kH + h) * kN + c * kBkt;
      __builtin_prefetch(ibn + (tid - kBkt) * 16, 0, 3);
    }
    __syncthreads();   // also fences previous round's PV fragment reads

    // gather sorted K/V rows (b128), f32 -> f16; V stored transposed for P.V
    for (int e = tid; e < kBkt * kDh / 4; e += 256) {
      const int row = e >> 4, c4 = (e & 15) * 4;
      const int src = sidx[row];
      const size_t g = (size_t)(b * kN + src) * kD + h * kDh + c4;
      const float4 kv = *(const float4*)(key + g);
      const float4 vv = *(const float4*)(val + g);
      Ks[row][c4 + 0] = (_Float16)kv.x;
      Ks[row][c4 + 1] = (_Float16)kv.y;
      Ks[row][c4 + 2] = (_Float16)kv.z;
      Ks[row][c4 + 3] = (_Float16)kv.w;
      VsT[c4 + 0][row] = (_Float16)vv.x;
      VsT[c4 + 1][row] = (_Float16)vv.y;
      VsT[c4 + 2][row] = (_Float16)vv.z;
      VsT[c4 + 3][row] = (_Float16)vv.w;
    }
    __syncthreads();

    // S = Q . K^T : two tiles per wave sharing one B-fragment per k-chunk
    v8f s0 = {}, s1 = {};
#pragma unroll
    for (int kc = 0; kc < 2; ++kc) {
      v16h bb = load_bT_frag(Ks, tjq * 16, kc * 32, lhalf, ln);
      s0 = __builtin_amdgcn_wmma_f32_16x16x32_f16(false, qa[0][kc], false, bb,
                                                  (short)0, s0, false, false);
      s1 = __builtin_amdgcn_wmma_f32_16x16x32_f16(false, qa[1][kc], false, bb,
                                                  (short)0, s1, false, false);
    }
    // C/D layout: n = ln, m = vgpr + 8*lhalf
#pragma unroll
    for (int p = 0; p < 8; ++p) {
      Ss[tiq * 16 + lhalf * 8 + p][tjq * 16 + ln]       = s0[p];
      Ss[(tiq + 2) * 16 + lhalf * 8 + p][tjq * 16 + ln] = s1[p];
    }
    __syncthreads();

    // softmax over rows of S (scale = 1/sqrt(64) = 0.125).
    // 4 lanes per row (quad in a wave), 16 cols per lane, shfl_xor reductions.
    {
      const int srow = tid >> 2;
      const int qcol = (tid & 3) * 16;
      float ex[16];
      float mx = -3.402823466e38f;
#pragma unroll
      for (int i = 0; i < 16; ++i) mx = fmaxf(mx, Ss[srow][qcol + i]);
      mx = fmaxf(mx, __shfl_xor(mx, 1, 4));
      mx = fmaxf(mx, __shfl_xor(mx, 2, 4));
      mx *= 0.125f;
      float sum = 0.f;
#pragma unroll
      for (int i = 0; i < 16; ++i) {
        ex[i] = __expf(Ss[srow][qcol + i] * 0.125f - mx);
        sum += ex[i];
      }
      sum += __shfl_xor(sum, 1, 4);
      sum += __shfl_xor(sum, 2, 4);
      const float inv = 1.f / sum;
#pragma unroll
      for (int i = 0; i < 16; ++i) {
        const float pnorm = ex[i] * inv;
        Ps[srow][qcol + i] = (_Float16)pnorm;
        Aacc[srow][qcol + i] += pnorm;      // sum over rounds (div later)
      }
    }
    __syncthreads();

    // O += P . V : both tiles share one B-fragment per k-chunk;
    // accumulation over rounds rides in the WMMA C operand.
#pragma unroll
    for (int kc = 0; kc < 2; ++kc) {
      const int k0 = kc * 32;
      v16h bb = load_bT_frag(VsT, tjq * 16, k0, lhalf, ln);
      v16h a0 = load_a_frag(Ps, tiq * 16, k0, lhalf, ln);
      oacc0 = __builtin_amdgcn_wmma_f32_16x16x32_f16(false, a0, false, bb,
                                                     (short)0, oacc0, false, false);
      v16h a1 = load_a_frag(Ps, (tiq + 2) * 16, k0, lhalf, ln);
      oacc1 = __builtin_amdgcn_wmma_f32_16x16x32_f16(false, a1, false, bb,
                                                     (short)0, oacc1, false, false);
    }
  }
  __syncthreads();

  // out[b, c*64+s, h*64+dd] = (1/R) * sum_r O_r  -- disjoint per block, plain stores
#pragma unroll
  for (int p = 0; p < 8; ++p) {
    const int s0r = tiq * 16 + lhalf * 8 + p;
    const int s1r = (tiq + 2) * 16 + lhalf * 8 + p;
    const int dd  = tjq * 16 + ln;
    out[(size_t)(b * kN + c * kBkt + s0r) * kD + h * kDh + dd] = oacc0[p] * (1.0f / kR);
    out[(size_t)(b * kN + c * kBkt + s1r) * kD + h * kDh + dd] = oacc1[p] * (1.0f / kR);
  }

  // attn_mean[b,c,s,e] += Aacc / (R*H)   (atomics only across the H blocks)
  for (int e = tid; e < kBkt * kBkt; e += 256) {
    const int s = e >> 6, t = e & 63;
    atomicAdd(&attn_out[(((size_t)b * kC + c) * kBkt + s) * kBkt + t],
              Aacc[s][t] * (1.0f / (kR * kH)));
  }
}

// ---------------------------------------------------------------------------
// Launcher
// ---------------------------------------------------------------------------
extern "C" void kernel_launch(void* const* d_in, const int* in_sizes, int n_in,
                              void* d_out, int out_size, void* d_ws, size_t ws_size,
                              hipStream_t stream) {
  const float* q      = (const float*)d_in[0];
  const float* key    = (const float*)d_in[1];
  const float* val    = (const float*)d_in[2];
  const float* valen  = (const float*)d_in[3];
  const float* proj   = (const float*)d_in[4];
  const float* vscale = (const float*)d_in[5];

  float* out      = (float*)d_out;                         // (B,N,D)
  float* attn_out = out + (size_t)kBatch * kN * kD;        // (B,C,64,64)

  int* codes = (int*)d_ws;                                 // B*R*H*N ints (2 MB)
  int* idx   = codes + (size_t)kBatch * kR * kH * kN;      // B*R*H*N ints (2 MB)

  const int attn_elems = kBatch * kC * kBkt * kBkt;
  lsh_zero_kernel<<<(attn_elems + 255) / 256, 256, 0, stream>>>(attn_out, attn_elems);

  lsh_codes_kernel<<<kBatch * kR * kH, 256, 0, stream>>>(q, proj, valen, vscale, codes);

  lsh_sort_kernel<<<kBatch * kR * kH, 32, 0, stream>>>(codes, idx, valen, vscale);

  lsh_attn_kernel<<<kBatch * kH * kC, 256, 0, stream>>>(q, key, val, idx, out, attn_out);
}
